// FMMLayer_36206574305291
// MI455X (gfx1250) — matrix operand order repaired and checked by reading, not compiled, not run on hardware
//
#include <hip/hip_runtime.h>
#include <hip/hip_bf16.h>
#include <math.h>

// ---------------------------------------------------------------------------
// MI455X (gfx1250) FMM layer:
//   out[b,n,o] = sin(fs[b,o] * (sum_i W[b,o,i]*x[b,n,i]) + ph[b,o])
//   W[b] = weight * (sigmoid(lowrank(freq)/sqrt(rank)) + 0.5)
//
// Roofline: 137 GFLOP vs 2.15 GB mandatory HBM traffic (x in + out out)
// => 64 FLOP/B => bf16 WMMA pipe (v_wmma_f32_16x16x32_bf16) keeps us at the
// 23.3 TB/s memory floor (~92 us). x converted fp32->bf16 in-register after
// the mandatory fp32 read; W pre-modulated to bf16 once (2 MiB, L2-resident,
// reused 65536x per batch).
//
// v2 changes vs v1 (from disasm feedback):
//  - all 4 B fragments per K-step loaded into distinct regs up-front so the
//    8 global_load_b128 clause together and overlap with the WMMA stream
//    (v1 reused one reg pair -> load/wait/wmma serialization)
//  - double-buffered LDS x-tile: one barrier per K-step instead of two;
//    next chunk's fp32 load + cvt overlaps current chunk's WMMAs
// ---------------------------------------------------------------------------

typedef __attribute__((ext_vector_type(16))) __bf16 v16bf;
typedef __attribute__((ext_vector_type(8)))  __bf16 v8bf;
typedef __attribute__((ext_vector_type(8)))  float  v8f;

#define B_SZ  16
#define N_SZ  65536
#define CIN   256
#define COUT  256
#define RANK  10

// ---------------------------------------------------------------------------
// Kernel 1 (tiny): per-batch modulated weight, bf16, layout [B][COUT][CIN]
// ---------------------------------------------------------------------------
__global__ void fmm_modulate(const float* __restrict__ freq,
                             const float* __restrict__ weight,
                             __bf16* __restrict__ Wmod) {
    const int idx = blockIdx.x * blockDim.x + threadIdx.x;  // b*65536 + o*256 + k
    const int b = idx >> 16;
    const int o = (idx >> 8) & 255;
    const int k = idx & 255;
    const float* fb = freq + (size_t)b * (RANK * (CIN + COUT));
    float acc = 0.f;
#pragma unroll
    for (int r = 0; r < RANK; ++r)
        acc += fb[o * RANK + r] * fb[COUT * RANK + r * CIN + k];
    acc *= 0.31622776601683794f;                 // 1/sqrt(RANK)
    const float sig = 1.0f / (1.0f + __expf(-acc));
    Wmod[idx] = (__bf16)(weight[o * CIN + k] * (sig + 0.5f));
}

// ---------------------------------------------------------------------------
// Kernel 2: batched GEMM (bf16 WMMA, f32 accum) fused with sin epilogue.
// Block = 256 threads (8 wave32), tile = 64 rows(n) x 256 cols(o).
// Wave (mw,ow) in 2x4 grid computes 32x64 = 2x4 WMMA 16x16 tiles.
// ---------------------------------------------------------------------------
__global__ __launch_bounds__(256, 2) void fmm_gemm_sin(
    const float*  __restrict__ x,           // [B][N][CIN] fp32
    const __bf16* __restrict__ Wmod,        // [B][COUT][CIN] bf16
    const float*  __restrict__ phase_shift, // [B][2*COUT] fp32
    float*        __restrict__ out)         // [B][N][COUT] fp32
{
    // double-buffered 64x32 bf16 x-tile: 2 x 4 KiB
    __shared__ __attribute__((aligned(32))) __bf16 sX[2][64 * 32];

    const int b    = blockIdx.y;
    const int n0   = blockIdx.x * 64;
    const int tid  = threadIdx.x;
    const int lane = tid & 31;
    const int wid  = tid >> 5;     // 0..7
    const int mw   = wid >> 2;     // 0..1 -> rows mw*32
    const int ow   = wid & 3;      // 0..3 -> cols ow*64
    const int hf   = lane >> 4;    // which 16-lane half
    const int l16  = lane & 15;

    const float*  xb = x    + ((size_t)b * N_SZ + n0) * CIN;
    const __bf16* Wb = Wmod +  (size_t)b * COUT * CIN;

    v8f acc[2][4];
#pragma unroll
    for (int mt = 0; mt < 2; ++mt)
#pragma unroll
        for (int ot = 0; ot < 4; ++ot)
#pragma unroll
            for (int e = 0; e < 8; ++e) acc[mt][ot][e] = 0.0f;

    // staging assignment: 4 threads per row, 8 contiguous k each
    const int srow = tid >> 2;
    const int scol = (tid & 3) * 8;
    const float* sbase = xb + (size_t)srow * CIN + scol;

    // ---- prologue: stage chunk k0=0 into buffer 0 ----
    {
        float4 f0 = *(const float4*)(sbase);
        float4 f1 = *(const float4*)(sbase + 4);
        v8bf pk;
        pk[0] = (__bf16)f0.x; pk[1] = (__bf16)f0.y;
        pk[2] = (__bf16)f0.z; pk[3] = (__bf16)f0.w;
        pk[4] = (__bf16)f1.x; pk[5] = (__bf16)f1.y;
        pk[6] = (__bf16)f1.z; pk[7] = (__bf16)f1.w;
        *(v8bf*)&sX[0][srow * 32 + scol] = pk;                  // ds_store_b128
        __builtin_prefetch(sbase + 32, 0, 3);                   // global_prefetch_b8
    }

#pragma unroll 1
    for (int k0 = 0; k0 < CIN; k0 += 32) {
        const int cur = (k0 >> 5) & 1;
        __syncthreads();   // staging of `cur` complete; prev readers done w/ `cur^1`

        // ---- A fragments from LDS: contiguous k order (same as B) ----
        // lanes 0-15:  row=l16, K = {0..15} of this 32-chunk
        // lanes 16-31: row=l16, K = {16..31}
        v16bf afrag[2];
#pragma unroll
        for (int mt = 0; mt < 2; ++mt) {
            const int row = mw * 32 + mt * 16 + l16;
            v8bf lo = *(const v8bf*)&sX[cur][row * 32 + hf * 16];
            v8bf hi = *(const v8bf*)&sX[cur][row * 32 + hf * 16 + 8];
#pragma unroll
            for (int e = 0; e < 8; ++e) { afrag[mt][e] = lo[e]; afrag[mt][8 + e] = hi[e]; }
        }

        // ---- all 4 B fragments up-front (distinct regs -> loads clause) ----
        // B layout: lane = N column (l16), halves split K 0..15 / 16..31,
        // elements contiguous in K -> one 32B vector load per tile.
        v16bf bfrag[4];
#pragma unroll
        for (int ot = 0; ot < 4; ++ot) {
            const int o = ow * 64 + ot * 16 + l16;
            bfrag[ot] = *(const v16bf*)(Wb + (size_t)o * CIN + k0 + hf * 16);
        }

        // ---- stage next chunk into the other buffer (overlaps WMMAs) ----
        if (k0 + 32 < CIN) {
            const float* src = sbase + k0 + 32;
            float4 f0 = *(const float4*)(src);
            float4 f1 = *(const float4*)(src + 4);
            v8bf pk;
            pk[0] = (__bf16)f0.x; pk[1] = (__bf16)f0.y;
            pk[2] = (__bf16)f0.z; pk[3] = (__bf16)f0.w;
            pk[4] = (__bf16)f1.x; pk[5] = (__bf16)f1.y;
            pk[6] = (__bf16)f1.z; pk[7] = (__bf16)f1.w;
            *(v8bf*)&sX[cur ^ 1][srow * 32 + scol] = pk;
            if (k0 + 64 < CIN)
                __builtin_prefetch(src + 32, 0, 3);             // next-next chunk
        }

        // ---- 8x WMMA ----
#pragma unroll
        for (int ot = 0; ot < 4; ++ot)
#pragma unroll
            for (int mt = 0; mt < 2; ++mt)
                acc[mt][ot] = __builtin_amdgcn_wmma_f32_16x16x32_bf16(
                    /*neg_a=*/false, afrag[mt],
                    /*neg_b=*/false, bfrag[ot],
                    /*c_mod=*/(short)0, acc[mt][ot],
                    /*reuse_a=*/false, /*reuse_b=*/false);
    }

    // ---- fused sin epilogue; C/D layout: lane=N col, VGPR r = M row ----
    const float* ps = phase_shift + (size_t)b * (2 * COUT);
#pragma unroll
    for (int ot = 0; ot < 4; ++ot) {
        const int o  = ow * 64 + ot * 16 + l16;
        const float fs = ps[o];
        const float ph = (ps[COUT + o] - 30.0f) * (1.0f / 15.0f);
#pragma unroll
        for (int mt = 0; mt < 2; ++mt) {
            const int rbase = mw * 32 + mt * 16 + hf * 8;
#pragma unroll
            for (int r = 0; r < 8; ++r) {
                const size_t n = (size_t)(n0 + rbase + r);
                out[((size_t)b * N_SZ + n) * COUT + o] =
                    __sinf(fs * acc[mt][ot][r] + ph);
            }
        }
    }
}

// ---------------------------------------------------------------------------
extern "C" void kernel_launch(void* const* d_in, const int* in_sizes, int n_in,
                              void* d_out, int out_size, void* d_ws, size_t ws_size,
                              hipStream_t stream) {
    const float* x      = (const float*)d_in[0];  // [16,65536,256]
    const float* freq   = (const float*)d_in[1];  // [16,5120]
    const float* phase  = (const float*)d_in[2];  // [16,512]
    const float* weight = (const float*)d_in[3];  // [256,256]
    __bf16* Wmod = (__bf16*)d_ws;                 // 2 MiB scratch

    // 1) modulated weights -> bf16 (one thread per element)
    fmm_modulate<<<dim3((B_SZ * COUT * CIN) / 256), dim3(256), 0, stream>>>(
        freq, weight, Wmod);

    // 2) batched GEMM + sin: 1024 n-tiles x 16 batches
    fmm_gemm_sin<<<dim3(N_SZ / 64, B_SZ), dim3(256), 0, stream>>>(
        x, Wmod, phase, (float*)d_out);
}
